// Network_85005992722489
// MI455X (gfx1250) — matrix operand order repaired and checked by reading, not compiled, not run on hardware
//
#include <hip/hip_runtime.h>
#include <hip/hip_bf16.h>
#include <math.h>

typedef _Float16 f16;
typedef _Float16 v16h __attribute__((ext_vector_type(16)));
typedef _Float16 v8h  __attribute__((ext_vector_type(8)));
typedef float    v8f  __attribute__((ext_vector_type(8)));
typedef unsigned int u32x4 __attribute__((ext_vector_type(4)));
typedef int          i32x8 __attribute__((ext_vector_type(8)));
typedef int          i32x4 __attribute__((ext_vector_type(4)));

#define HF 75
#define WF 120
#define NPOS (HF*WF)            // 9000
#define CIN 512
#define KTOT (CIN*9)            // 4608
#define KS32 (KTOT/32)          // 144
#define MTILES ((NPOS+15)/16)   // 563
#define PADH (HF+2)
#define PADW (WF+2)
#define NHEAD 64                // 18 cls + 36 bbox + 10 pad
#define KSH 16                  // 512/32
#define PRE_NMS 6000
#define POST_NMS 300
#define SORTN 8192
#define NPROP (NPOS*9)          // 81000
#define IMW 1920.0f
#define IMH 1200.0f

#if __has_builtin(__builtin_amdgcn_tensor_load_to_lds)
#define USE_TDM 1
#else
#define USE_TDM 0
#endif

static __device__ __forceinline__ v8f wmma16(v16h a, v16h b, v8f c) {
  return __builtin_amdgcn_wmma_f32_16x16x32_f16(false, a, false, b, (short)0, c,
                                                false, false);
}
static __device__ __forceinline__ v16h cat8(v8h lo, v8h hi) {
  v16h r;
#pragma unroll
  for (int j = 0; j < 8; ++j) { r[j] = lo[j]; r[8 + j] = hi[j]; }
  return r;
}

#if USE_TDM
// Issue one TDM descriptor: contiguous 4KB (512 x 8B) global -> LDS.
// D# group0/group1 bit packing per CDNA5 ISA 8.3/8.4 (1-D tile, data_size=3=8B).
// Toolchain here is the 6-arg clang-23 variant:
//   (u32x4 g0, i32x8 g1, i32x4, i32x4, i32x8, i32 cpol)
static __device__ __forceinline__ void tdm_load_4k(unsigned lds_off,
                                                   const void* gptr) {
  const unsigned long long ga = (unsigned long long)(uintptr_t)gptr;
  u32x4 g0;
  g0[0] = 1u;                                              // count=1 (user D#)
  g0[1] = lds_off;                                         // lds_addr (bytes)
  g0[2] = (unsigned)ga;                                    // global_addr[31:0]
  g0[3] = (unsigned)((ga >> 32) & 0x1FFFFFFu) | (2u << 30);// addr[56:32]|type=2
  i32x8 g1;
  g1[0] = (3 << 16);        // workgroup_mask=0, data_size=3 (8B), no flags
  g1[1] = (512 << 16);      // tensor_dim0[15:0]=512  (bits 79:48)
  g1[2] = (1 << 16);        // tensor_dim0[31:16]=0, tensor_dim1[15:0]=1
  g1[3] = (512 << 16);      // tensor_dim1[31:16]=0, tile_dim0=512
  g1[4] = 1;                // tile_dim1=1, tile_dim2=0
  g1[5] = 512;              // tensor_dim0_stride[31:0]=512
  g1[6] = (512 << 16);      // stride0[47:32]=0, tensor_dim1_stride[15:0]=512
  g1[7] = 0;                // stride1[47:16]=0
  const i32x4 z4 = {0, 0, 0, 0};
  const i32x8 z8 = {0, 0, 0, 0, 0, 0, 0, 0};
  __builtin_amdgcn_tensor_load_to_lds(g0, g1, z4, z4, z8, 0);
}
#endif

// ---- Kernel 1: NCHW f32 -> padded HWC f16 (77 x 122 x 512, zero border) ----
__global__ void k_pack(const float* __restrict__ x, f16* __restrict__ padIn) {
  const int total = PADH * PADW * CIN;
  for (int i = blockIdx.x * blockDim.x + threadIdx.x; i < total;
       i += gridDim.x * blockDim.x) {
    int c = i & (CIN - 1);
    int r = i >> 9;
    int xx = r % PADW, yy = r / PADW;
    float v = 0.f;
    if (yy >= 1 && yy <= HF && xx >= 1 && xx <= WF)
      v = x[((size_t)c * HF + (yy - 1)) * WF + (xx - 1)];
    padIn[i] = (f16)v;
  }
}

// ---- Kernel 2: conv weight OIHW f32 -> TDM-friendly fragment B f16 ----
// Layout: Bp[ks][wave][t][lane][16 halves]  (each (ks,wave) chunk = 4KB contig)
// n = (wave*4+t)*16 + lane%16 ; k = ks*32 + (lane/16)*16 + j ; k=(ky*3+kx)*512+c
__global__ void k_wconv(const float* __restrict__ w, f16* __restrict__ Bp) {
  const int total = KS32 * 8 * 4 * 32 * 16;
  for (int i = blockIdx.x * blockDim.x + threadIdx.x; i < total;
       i += gridDim.x * blockDim.x) {
    int j    = i & 15;
    int l    = (i >> 4) & 31;
    int t    = (i >> 9) & 3;
    int wv   = (i >> 11) & 7;
    int ks   = i >> 14;
    int k    = ks * 32 + (l >> 4) * 16 + j;
    int n    = (wv * 4 + t) * 16 + (l & 15);
    int c    = k & (CIN - 1);
    int dydx = k >> 9;
    int ky = dydx / 3, kx = dydx % 3;
    Bp[i] = (f16)w[(((size_t)n * CIN + c) * 3 + ky) * 3 + kx];
  }
}

// ---- Kernel 3: head weights (cls 18 + bbox 36, pad to 64) -> fragment B ----
__global__ void k_whead(const float* __restrict__ clsw,
                        const float* __restrict__ bbxw, f16* __restrict__ Bp) {
  const int total = 4 * KSH * 32 * 16;
  for (int i = blockIdx.x * blockDim.x + threadIdx.x; i < total;
       i += gridDim.x * blockDim.x) {
    int j  = i & 15;
    int l  = (i >> 4) & 31;
    int ks = (i >> 9) % KSH;
    int nt = (i >> 9) / KSH;
    int k  = ks * 32 + (l >> 4) * 16 + j;
    int n  = nt * 16 + (l & 15);
    float v = 0.f;
    if (n < 18)      v = clsw[(size_t)n * CIN + k];
    else if (n < 54) v = bbxw[(size_t)(n - 18) * CIN + k];
    Bp[i] = (f16)v;
  }
}

// ---- Kernel 4: conv3x3 + bias + ReLU as WMMA implicit GEMM ----
// block = 256 threads = 8 waves; wave: 16M x 64N (4 accums).
// A: register-prefetch pipelined into double-buffered LDS (1 barrier/iter).
// B: per-wave TDM async copy into double-buffered LDS, s_wait_tensorcnt sync.
__global__ __launch_bounds__(256) void k_conv_gemm(
    const f16* __restrict__ padIn, const f16* __restrict__ Bp,
    const float* __restrict__ bias, f16* __restrict__ rpn) {
  __shared__ f16 ldsA[2][16 * 32];
#if USE_TDM
  __shared__ f16 ldsB[2][8][4 * 32 * 16];  // 64KB
#endif
  const int mtile = blockIdx.x;
  const int tid = threadIdx.x;
  const int wave = tid >> 5, lane = tid & 31;
  const int g = lane >> 4, ln = lane & 15;
  v8f acc[4] = {v8f{}, v8f{}, v8f{}, v8f{}};

  // per-thread A staging coords: row am (0..15), dword (tid&15)
  const int am = tid >> 4;
  int p_ld = mtile * 16 + am;
  if (p_ld > NPOS - 1) p_ld = NPOS - 1;
  const int py = p_ld / WF, px = p_ld % WF;

  auto aload = [&](int ks) -> unsigned {
    const int dydx = ks >> 4;
    const int coff = (ks & 15) * 32;
    const int dy = dydx / 3, dx = dydx % 3;
    const unsigned* sp = (const unsigned*)(
        padIn + ((size_t)((py + dy) * PADW + (px + dx))) * CIN + coff);
    return sp[tid & 15];
  };

#if USE_TDM
  auto bissue = [&](int ks, int buf) {
    tdm_load_4k((unsigned)(uintptr_t)(&ldsB[buf][wave][0]),
                Bp + ((size_t)(ks * 8 + wave)) * 2048);
  };
  bissue(0, 0);
#endif
  unsigned areg = aload(0);

  for (int ks = 0; ks < KS32; ++ks) {
    const int cur = ks & 1;
    ((unsigned*)ldsA[cur])[tid] = areg;
    if (ks + 1 < KS32) {
#if USE_TDM
      bissue(ks + 1, cur ^ 1);
#endif
      areg = aload(ks + 1);
    }
    __syncthreads();
#if USE_TDM
    if (ks + 1 < KS32) __builtin_amdgcn_s_wait_tensorcnt(1);
    else               __builtin_amdgcn_s_wait_tensorcnt(0);
#endif
    const v8h lo = *(const v8h*)&ldsA[cur][ln * 32 + g * 16];
    const v8h hi = *(const v8h*)&ldsA[cur][ln * 32 + g * 16 + 8];
    const v16h afrag = cat8(lo, hi);
#pragma unroll
    for (int t = 0; t < 4; ++t) {
#if USE_TDM
      const v16h bfrag = *(const v16h*)&ldsB[cur][wave][(t * 32 + lane) * 16];
#else
      const v16h bfrag = *(const v16h*)(
          Bp + (((size_t)(ks * 8 + wave) * 4 + t) * 32 + lane) * 16);
#endif
      acc[t] = wmma16(afrag, bfrag, acc[t]);
    }
  }

  const bool full = (mtile < MTILES - 1);  // 9000 = 562*16 + 8
#pragma unroll
  for (int t = 0; t < 4; ++t) {
    const int n = (wave * 4 + t) * 16 + ln;
    const float bv = bias[n];
    if (full) {
#pragma unroll
      for (int r = 0; r < 8; ++r) {
        const int p = mtile * 16 + g * 8 + r;
        float v = acc[t][r] + bv;
        rpn[(size_t)p * CIN + n] = (f16)(v > 0.f ? v : 0.f);
      }
    } else {
#pragma unroll
      for (int r = 0; r < 8; ++r) {
        const int p = mtile * 16 + g * 8 + r;
        if (p < NPOS) {
          float v = acc[t][r] + bv;
          rpn[(size_t)p * CIN + n] = (f16)(v > 0.f ? v : 0.f);
        }
      }
    }
  }
}

// ---- Kernel 5: 1x1 head convs (cls+bbox fused, N=64) via WMMA ----
__global__ __launch_bounds__(128) void k_head_gemm(
    const f16* __restrict__ rpn, const f16* __restrict__ Bp,
    const float* __restrict__ clsb, const float* __restrict__ bbxb,
    float* __restrict__ headOut) {
  const int mtile = blockIdx.x;
  const int tid = threadIdx.x;
  const int wave = tid >> 5, lane = tid & 31;
  const int g = lane >> 4, ln = lane & 15;
  v8f acc = v8f{};

  int p = mtile * 16 + ln;
  if (p > NPOS - 1) p = NPOS - 1;
  const f16* arow = rpn + (size_t)p * CIN;

  for (int ks = 0; ks < KSH; ++ks) {
    const v8h lo = *(const v8h*)(arow + ks * 32 + g * 16);
    const v8h hi = *(const v8h*)(arow + ks * 32 + g * 16 + 8);
    const v16h afrag = cat8(lo, hi);
    const v16h bfrag =
        *(const v16h*)(Bp + (((size_t)wave * KSH + ks) * 32 + lane) * 16);
    acc = wmma16(afrag, bfrag, acc);
  }

  const int n = wave * 16 + ln;
  float bv = 0.f;
  if (n < 18) bv = clsb[n];
  else if (n < 54) bv = bbxb[n - 18];
  if (mtile < MTILES - 1) {
#pragma unroll
    for (int r = 0; r < 8; ++r)
      headOut[(size_t)(mtile * 16 + g * 8 + r) * NHEAD + n] = acc[r] + bv;
  } else {
#pragma unroll
    for (int r = 0; r < 8; ++r) {
      const int pp = mtile * 16 + g * 8 + r;
      if (pp < NPOS) headOut[(size_t)pp * NHEAD + n] = acc[r] + bv;
    }
  }
}

// ---- Kernel 6: softmax fg prob + anchor decode + clip ----
__global__ void k_decode(const float* __restrict__ headOut,
                         float* __restrict__ boxes, float* __restrict__ scores) {
  for (int i = blockIdx.x * blockDim.x + threadIdx.x; i < NPROP;
       i += gridDim.x * blockDim.x) {
    const int h = i / (WF * 9);
    const int rem = i % (WF * 9);
    const int w = rem / 9;
    const int a = rem % 9;
    const int p = h * WF + w;
    const float* row = headOut + (size_t)p * NHEAD;
    const float s0 = row[a], s1 = row[9 + a];
    const float sc = 1.f / (1.f + expf(s0 - s1));
    const float ddx = row[18 + 4 * a + 0];
    const float ddy = row[18 + 4 * a + 1];
    const float ddw = row[18 + 4 * a + 2];
    const float ddh = row[18 + 4 * a + 3];

    const int ri = a / 3, si = a % 3;
    const float ratio = (ri == 0) ? 0.5f : ((ri == 1) ? 1.0f : 2.0f);
    const float scale = (si == 0) ? 8.f : ((si == 1) ? 16.f : 32.f);
    const float wr = roundf(sqrtf(256.f / ratio));
    const float hr = roundf(wr * ratio);
    const float aws = wr * scale, ahs = hr * scale;
    const float sx = (float)w * 16.f, sy = (float)h * 16.f;
    const float ax1 = 7.5f - 0.5f * (aws - 1.f) + sx;
    const float ay1 = 7.5f - 0.5f * (ahs - 1.f) + sy;
    const float ax2 = 7.5f + 0.5f * (aws - 1.f) + sx;
    const float ay2 = 7.5f + 0.5f * (ahs - 1.f) + sy;
    const float aw = ax2 - ax1 + 1.f, ah = ay2 - ay1 + 1.f;
    const float acx = ax1 + 0.5f * aw, acy = ay1 + 0.5f * ah;

    const float pcx = ddx * aw + acx;
    const float pcy = ddy * ah + acy;
    const float pw = expf(ddw) * aw;
    const float ph = expf(ddh) * ah;
    float x1 = pcx - 0.5f * pw, y1 = pcy - 0.5f * ph;
    float x2 = pcx + 0.5f * pw, y2 = pcy + 0.5f * ph;
    x1 = fminf(fmaxf(x1, 0.f), IMW - 1.f);
    y1 = fminf(fmaxf(y1, 0.f), IMH - 1.f);
    x2 = fminf(fmaxf(x2, 0.f), IMW - 1.f);
    y2 = fminf(fmaxf(y2, 0.f), IMH - 1.f);
    boxes[(size_t)i * 4 + 0] = x1;
    boxes[(size_t)i * 4 + 1] = y1;
    boxes[(size_t)i * 4 + 2] = x2;
    boxes[(size_t)i * 4 + 3] = y2;
    scores[i] = sc;
  }
}

// ---- Kernel 7: reset scratch for top-k selection ----
__global__ void k_init(int* hist, int* meta, float* selS, int* selI) {
  const int i = blockIdx.x * blockDim.x + threadIdx.x;
  if (i < 1024) hist[i] = 0;
  if (i < 4) meta[i] = 0;
  if (i < SORTN) { selS[i] = -1e30f; selI[i] = 0; }
}

__global__ void k_hist(const float* __restrict__ scores, int* __restrict__ hist) {
  for (int i = blockIdx.x * blockDim.x + threadIdx.x; i < NPROP;
       i += gridDim.x * blockDim.x) {
    int b = (int)(scores[i] * 1024.f);
    b = b < 0 ? 0 : (b > 1023 ? 1023 : b);
    atomicAdd(&hist[b], 1);
  }
}

__global__ void k_thresh(const int* __restrict__ hist, int* __restrict__ meta) {
  if (threadIdx.x == 0 && blockIdx.x == 0) {
    int run = 0, thr = 0, above = 0, found = 0;
    for (int i = 1023; i >= 0; --i) {
      const int c = hist[i];
      if (run + c >= PRE_NMS) { thr = i; above = run; found = 1; break; }
      run += c;
    }
    if (!found) { thr = 0; above = run; }
    meta[0] = thr;
    meta[1] = above;
  }
}

__global__ void k_compact(const float* __restrict__ scores,
                          const int* __restrict__ meta_ro, int* __restrict__ meta,
                          float* __restrict__ selS, int* __restrict__ selI) {
  const int thr = meta_ro[0];
  const int above = meta_ro[1];
  for (int i = blockIdx.x * blockDim.x + threadIdx.x; i < NPROP;
       i += gridDim.x * blockDim.x) {
    const float s = scores[i];
    int b = (int)(s * 1024.f);
    b = b < 0 ? 0 : (b > 1023 ? 1023 : b);
    if (b > thr) {
      const int pos = atomicAdd(&meta[2], 1);
      if (pos < SORTN) { selS[pos] = s; selI[pos] = i; }
    } else if (b == thr) {
      const int pos = atomicAdd(&meta[3], 1);
      const int slot = above + pos;
      if (slot < PRE_NMS) { selS[slot] = s; selI[slot] = i; }
    }
  }
}

// ---- Kernel 8: single-block bitonic sort (descending) + gather boxes ----
__global__ __launch_bounds__(1024) void k_sort(const float* __restrict__ selS,
                                               const int* __restrict__ selI,
                                               const float* __restrict__ boxes,
                                               float* __restrict__ boxesK) {
  __shared__ float ss[SORTN];
  __shared__ int si[SORTN];
  const int tid = threadIdx.x;
  for (int i = tid; i < SORTN; i += 1024) { ss[i] = selS[i]; si[i] = selI[i]; }
  __syncthreads();
  for (int k = 2; k <= SORTN; k <<= 1) {
    for (int j = k >> 1; j > 0; j >>= 1) {
      for (int i = tid; i < SORTN; i += 1024) {
        const int p = i ^ j;
        if (p > i) {
          const bool up = ((i & k) == 0);  // descending in "up" regions
          const float vi = ss[i], vp = ss[p];
          const bool sw = up ? (vi < vp) : (vi > vp);
          if (sw) {
            ss[i] = vp; ss[p] = vi;
            const int t = si[i]; si[i] = si[p]; si[p] = t;
          }
        }
      }
      __syncthreads();
    }
  }
  for (int t = tid; t < PRE_NMS; t += 1024) {
    const int id = si[t];
    boxesK[(size_t)t * 4 + 0] = boxes[(size_t)id * 4 + 0];
    boxesK[(size_t)t * 4 + 1] = boxes[(size_t)id * 4 + 1];
    boxesK[(size_t)t * 4 + 2] = boxes[(size_t)id * 4 + 2];
    boxesK[(size_t)t * 4 + 3] = boxes[(size_t)id * 4 + 3];
  }
}

// ---- Kernel 9: greedy NMS (sorted order) + first-300 ROI normalization ----
__global__ __launch_bounds__(1024) void k_nms(const float* __restrict__ boxesK,
                                              float* __restrict__ roiBn) {
  __shared__ unsigned char keep[PRE_NMS];
  __shared__ int roiIdx[POST_NMS];
  const int tid = threadIdx.x;
  for (int i = tid; i < PRE_NMS; i += 1024) keep[i] = 1;
  __syncthreads();
  for (int i = 0; i < PRE_NMS; ++i) {
    __syncthreads();
    if (!keep[i]) continue;  // uniform: keep[i] stable at this point
    const float x1 = boxesK[(size_t)i * 4 + 0];
    const float y1 = boxesK[(size_t)i * 4 + 1];
    const float x2 = boxesK[(size_t)i * 4 + 2];
    const float y2 = boxesK[(size_t)i * 4 + 3];
    const float ai = (x2 - x1 + 1.f) * (y2 - y1 + 1.f);
    for (int j = i + 1 + tid; j < PRE_NMS; j += 1024) {
      if (!keep[j]) continue;
      const float bx1 = boxesK[(size_t)j * 4 + 0];
      const float by1 = boxesK[(size_t)j * 4 + 1];
      const float bx2 = boxesK[(size_t)j * 4 + 2];
      const float by2 = boxesK[(size_t)j * 4 + 3];
      const float aj = (bx2 - bx1 + 1.f) * (by2 - by1 + 1.f);
      const float xx1 = fmaxf(x1, bx1), yy1 = fmaxf(y1, by1);
      const float xx2 = fminf(x2, bx2), yy2 = fminf(y2, by2);
      const float iw = fmaxf(xx2 - xx1 + 1.f, 0.f);
      const float ih = fmaxf(yy2 - yy1 + 1.f, 0.f);
      const float inter = iw * ih;
      if (inter / (ai + aj - inter) > 0.7f) keep[j] = 0;
    }
  }
  __syncthreads();
  if (tid == 0) {
    int cnt = 0;
    for (int i = 0; i < PRE_NMS && cnt < POST_NMS; ++i)
      if (keep[i]) roiIdx[cnt++] = i;
    for (int i = 0; i < PRE_NMS && cnt < POST_NMS; ++i)
      if (!keep[i]) roiIdx[cnt++] = i;
  }
  __syncthreads();
  for (int r = tid; r < POST_NMS; r += 1024) {
    const int id = roiIdx[r];
    const float x1 = boxesK[(size_t)id * 4 + 0];
    const float y1 = boxesK[(size_t)id * 4 + 1];
    const float x2 = boxesK[(size_t)id * 4 + 2];
    const float y2 = boxesK[(size_t)id * 4 + 3];
    roiBn[(size_t)r * 4 + 0] = y1 / 16.f / (float)(HF - 1);
    roiBn[(size_t)r * 4 + 1] = x1 / 16.f / (float)(WF - 1);
    roiBn[(size_t)r * 4 + 2] = y2 / 16.f / (float)(HF - 1);
    roiBn[(size_t)r * 4 + 3] = x2 / 16.f / (float)(WF - 1);
  }
}

// ---- Kernel 10: 14x14 bilinear crop + 2x2 maxpool -> (300,512,7,7) ----
__global__ void k_roipool(const float* __restrict__ feat,
                          const float* __restrict__ roiBn,
                          float* __restrict__ out) {
  const int total = POST_NMS * CIN * 49;
  for (int i = blockIdx.x * blockDim.x + threadIdx.x; i < total;
       i += gridDim.x * blockDim.x) {
    const int px = i % 7;
    const int py = (i / 7) % 7;
    const int c = (i / 49) % CIN;
    const int n = i / (49 * CIN);
    const float y1 = roiBn[(size_t)n * 4 + 0];
    const float x1 = roiBn[(size_t)n * 4 + 1];
    const float y2 = roiBn[(size_t)n * 4 + 2];
    const float x2 = roiBn[(size_t)n * 4 + 3];
    const float* fc = feat + (size_t)c * HF * WF;
    float m = -1e30f;
#pragma unroll
    for (int sy = 0; sy < 2; ++sy) {
#pragma unroll
      for (int sx = 0; sx < 2; ++sx) {
        const float ty = (float)(2 * py + sy) / 13.f;
        const float tx = (float)(2 * px + sx) / 13.f;
        const float ys = (y1 + ty * (y2 - y1)) * (float)(HF - 1);
        const float xs = (x1 + tx * (x2 - x1)) * (float)(WF - 1);
        const float yf = floorf(ys), xf = floorf(xs);
        const float wy = ys - yf, wx = xs - xf;
        int yl = (int)yf; yl = yl < 0 ? 0 : (yl > HF - 1 ? HF - 1 : yl);
        int xl = (int)xf; xl = xl < 0 ? 0 : (xl > WF - 1 ? WF - 1 : xl);
        const int yh = yl + 1 > HF - 1 ? HF - 1 : yl + 1;
        const int xh = xl + 1 > WF - 1 ? WF - 1 : xl + 1;
        const float tll = fc[yl * WF + xl], tlh = fc[yl * WF + xh];
        const float thl = fc[yh * WF + xl], thh = fc[yh * WF + xh];
        const float top = tll * (1.f - wx) + tlh * wx;
        const float bot = thl * (1.f - wx) + thh * wx;
        const float v = top * (1.f - wy) + bot * wy;
        m = fmaxf(m, v);
      }
    }
    out[i] = m;
  }
}

extern "C" void kernel_launch(void* const* d_in, const int* in_sizes, int n_in,
                              void* d_out, int out_size, void* d_ws,
                              size_t ws_size, hipStream_t stream) {
  const float* net_conv = (const float*)d_in[0];  // (1,512,75,120)
  const float* rpn_w    = (const float*)d_in[1];  // (512,512,3,3)
  const float* rpn_b    = (const float*)d_in[2];  // (512)
  const float* cls_w    = (const float*)d_in[3];  // (18,512,1,1)
  const float* cls_b    = (const float*)d_in[4];  // (18)
  const float* bbox_w   = (const float*)d_in[5];  // (36,512,1,1)
  const float* bbox_b   = (const float*)d_in[6];  // (36)
  float* out = (float*)d_out;                     // (300,512,7,7)

  size_t off = 0;
  auto alloc = [&](size_t bytes) -> void* {
    off = (off + 255) & ~(size_t)255;
    void* p = (char*)d_ws + off;
    off += bytes;
    return p;
  };
  f16*   padIn   = (f16*)alloc((size_t)PADH * PADW * CIN * 2);
  f16*   BpConv  = (f16*)alloc((size_t)KS32 * 8 * 4 * 32 * 16 * 2);
  f16*   BpHead  = (f16*)alloc((size_t)4 * KSH * 32 * 16 * 2);
  f16*   rpn     = (f16*)alloc((size_t)NPOS * CIN * 2);
  float* headOut = (float*)alloc((size_t)NPOS * NHEAD * 4);
  float* pBoxes  = (float*)alloc((size_t)NPROP * 4 * 4);
  float* pScores = (float*)alloc((size_t)NPROP * 4);
  int*   hist    = (int*)alloc(1024 * 4);
  int*   meta    = (int*)alloc(4 * 4);
  float* selS    = (float*)alloc((size_t)SORTN * 4);
  int*   selI    = (int*)alloc((size_t)SORTN * 4);
  float* boxesK  = (float*)alloc((size_t)PRE_NMS * 4 * 4);
  float* roiBn   = (float*)alloc((size_t)POST_NMS * 4 * 4);
  (void)ws_size; (void)n_in; (void)in_sizes; (void)out_size;

  k_pack<<<(PADH * PADW * CIN + 255) / 256, 256, 0, stream>>>(net_conv, padIn);
  k_wconv<<<(KS32 * 8 * 4 * 32 * 16 + 255) / 256, 256, 0, stream>>>(rpn_w,
                                                                    BpConv);
  k_whead<<<(4 * KSH * 32 * 16 + 255) / 256, 256, 0, stream>>>(cls_w, bbox_w,
                                                               BpHead);
  k_conv_gemm<<<MTILES, 256, 0, stream>>>(padIn, BpConv, rpn_b, rpn);
  k_head_gemm<<<MTILES, 128, 0, stream>>>(rpn, BpHead, cls_b, bbox_b, headOut);
  k_decode<<<(NPROP + 255) / 256, 256, 0, stream>>>(headOut, pBoxes, pScores);
  k_init<<<(SORTN + 255) / 256, 256, 0, stream>>>(hist, meta, selS, selI);
  k_hist<<<(NPROP + 255) / 256, 256, 0, stream>>>(pScores, hist);
  k_thresh<<<1, 32, 0, stream>>>(hist, meta);
  k_compact<<<(NPROP + 255) / 256, 256, 0, stream>>>(pScores, meta, meta, selS,
                                                     selI);
  k_sort<<<1, 1024, 0, stream>>>(selS, selI, pBoxes, boxesK);
  k_nms<<<1, 1024, 0, stream>>>(boxesK, roiBn);
  k_roipool<<<(POST_NMS * CIN * 49 + 255) / 256, 256, 0, stream>>>(net_conv,
                                                                   roiBn, out);
}
